// MonotonicGruCell_21973052686473
// MI455X (gfx1250) — matrix operand order repaired and checked by reading, not compiled, not run on hardware
//
#include <hip/hip_runtime.h>

typedef __attribute__((ext_vector_type(16))) __bf16 v16bf;
typedef __attribute__((ext_vector_type(8)))  __bf16 v8bf;
typedef __attribute__((ext_vector_type(8)))  float  v8f;
typedef __attribute__((ext_vector_type(4)))  unsigned int v4u;
typedef __attribute__((ext_vector_type(8)))  int  v8i;
typedef __attribute__((ext_vector_type(4)))  int  v4i;

#define B_DIM 16384
#define H_DIM 1024
#define TM 128      // block tile M
#define TN 256      // block tile N
#define KB 32       // K step (one bf16 WMMA K)
#define PITCH 40    // LDS row pitch in bf16 elems (80 B = 64 B data + 16 B pad)

// Tensor Data Mover path: probe-verified builtin, arity differs per toolchain.
#if defined(__has_builtin)
#  if __has_builtin(__builtin_amdgcn_tensor_load_to_lds) && \
      __has_builtin(__builtin_amdgcn_s_wait_tensorcnt)
#    define USE_TDM 1
#  endif
#endif

union FragBF { v16bf v; v8bf h[2]; };

__device__ __forceinline__ float sigmoidf(float x) {
    return 1.0f / (1.0f + __expf(-x));
}

#ifdef USE_TDM
__device__ __forceinline__ unsigned lds_addr_of(const void* p) {
    return (unsigned)(unsigned long long)
        (__attribute__((address_space(3))) const void*)p;
}

// Issue one 2D TDM tile load: rows x 32 bf16 elements, row stride = K elems,
// LDS destination padded 16B per 64B row (pad_interval=16 dw, pad_amount=4 dw).
__device__ __forceinline__ void tdm_load_tile(const __bf16* gsrc,
                                              unsigned lds_byte_addr,
                                              int K_elems, int rows,
                                              int tensor_rows) {
    unsigned long long ga = (unsigned long long)gsrc;
    v4u g0;
    g0[0] = 1u;                                       // count=1, user mode
    g0[1] = lds_byte_addr;                            // lds_addr
    g0[2] = (unsigned)(ga & 0xFFFFFFFFu);             // global_addr[31:0]
    g0[3] = (unsigned)((ga >> 32) & 0x01FFFFFFu)      // global_addr[56:32]
          | (2u << 30);                               // type=2 (image)
    v8i g1;
    g1[0] = (int)((1u << 16)                          // data_size = 2 bytes
                | (1u << 20)                          // pad_enable
                | (3u << 22)                          // pad_interval: 16 dwords
                | (3u << 25));                        // pad_amount: 4 dwords
    g1[1] = (int)(((unsigned)K_elems & 0xFFFFu) << 16);          // dim0[15:0]
    g1[2] = (int)((((unsigned)K_elems >> 16) & 0xFFFFu)          // dim0[31:16]
                | (((unsigned)tensor_rows & 0xFFFFu) << 16));    // dim1[15:0]
    g1[3] = (int)((((unsigned)tensor_rows >> 16) & 0xFFFFu)      // dim1[31:16]
                | (32u << 16));                                  // tile_dim0=32
    g1[4] = rows & 0xFFFF;                            // tile_dim1 (tile_dim2=0)
    g1[5] = K_elems;                                  // tensor_dim0_stride[31:0]
    g1[6] = 0;                                        // stride hi / dim1_stride lo
    g1[7] = 0;
    v4i z4 = (v4i)0;
#if __clang_major__ >= 23
    v8i z8 = (v8i)0;
    __builtin_amdgcn_tensor_load_to_lds(g0, g1, z4, z4, z8, 0);
#else
    __builtin_amdgcn_tensor_load_to_lds(g0, g1, z4, z4, 0);
#endif
}
#endif // USE_TDM

// ---------------------------------------------------------------------------
// Prep kernels: f32 -> bf16 conversions + fused-weight construction
// ---------------------------------------------------------------------------
__global__ void prep_xcat(const float* __restrict__ x,
                          const float* __restrict__ hidden,
                          __bf16* __restrict__ Xcat) {
    size_t i = (size_t)blockIdx.x * 256 + threadIdx.x;   // over B*2048
    int b = (int)(i >> 11);
    int c = (int)(i & 2047);
    float v = (c < H_DIM) ? x[(size_t)b * H_DIM + c]
                          : hidden[(size_t)b * H_DIM + (c - H_DIM)];
    Xcat[i] = (__bf16)v;
}

__global__ void prep_wcat(const float* __restrict__ Wi,
                          const float* __restrict__ Wh,
                          __bf16* __restrict__ Wcat) {
    size_t i = (size_t)blockIdx.x * 256 + threadIdx.x;   // over 3072*2048
    int r = (int)(i >> 11);
    int c = (int)(i & 2047);
    float v;
    if (c < H_DIM)          v = Wi[(size_t)r * H_DIM + c];
    else if (r < 2 * H_DIM) v = Wh[(size_t)r * H_DIM + (c - H_DIM)];
    else                    v = 0.0f;   // n-gate row: hidden half masked out
    Wcat[i] = (__bf16)v;
}

__global__ void prep_whn(const float* __restrict__ Wh, __bf16* __restrict__ Whn) {
    size_t i = (size_t)blockIdx.x * 256 + threadIdx.x;   // over 1024*1024
    Whn[i] = (__bf16)Wh[(size_t)2 * H_DIM * H_DIM + i];
}

__global__ void prep_bcat(const float* __restrict__ bi,
                          const float* __restrict__ bh,
                          float* __restrict__ bcat) {
    int i = blockIdx.x * 256 + threadIdx.x;              // over 3072
    bcat[i] = bi[i] + ((i < 2 * H_DIM) ? bh[i] : 0.0f);
}

// ---------------------------------------------------------------------------
// Fused bf16 WMMA GEMM:  C = A(MxK) * Bw(NxK)^T + bias, fused GRU epilogue.
// TDM-fed double-buffered LDS when available; sync staging fallback.
// MODE 0: N=3072, emits rgh(bf16), inputgate, i_n.
// MODE 1: N=1024, emits hy = hidden + (1-hidden)*ig*sigmoid(i_n + h_n).
// ---------------------------------------------------------------------------
template <int MODE>
__global__ __launch_bounds__(256) void gru_gemm(
    const __bf16* __restrict__ A,      // M x K row-major (bf16)
    const __bf16* __restrict__ Bw,     // N x K row-major (bf16 weights)
    const float*  __restrict__ bias,   // N
    int K, int Nrows,
    const float*  __restrict__ hidden, // B x H (f32)
    __bf16*       __restrict__ rgh,    // MODE0 out: hidden * resetgate (bf16)
    float*        __restrict__ ig,     // MODE0 out / MODE1 in: inputgate
    float*        __restrict__ inb,    // MODE0 out / MODE1 in: i_n
    float*        __restrict__ outp)   // MODE1 out: hy
{
    __shared__ __align__(16) __bf16 sA[2][TM * PITCH];
    __shared__ __align__(16) __bf16 sB[2][TN * PITCH];

    const int tid  = threadIdx.x;
    const int lane = tid & 31;
    const int wave = tid >> 5;
    const int m0   = (wave >> 2) * 64;
    const int n0   = (wave & 3) * 64;
    const int bm   = blockIdx.x * TM;
    const int bn   = blockIdx.y * TN;

    const int half = lane >> 4;          // 0/1
    const int lrow = lane & 15;

    v8f acc[4][4];
#pragma unroll
    for (int i = 0; i < 4; ++i)
#pragma unroll
        for (int j = 0; j < 4; ++j)
            acc[i][j] = (v8f)0.0f;

#ifdef USE_TDM
    // ---- TDM double-buffered main loop --------------------------------
    if (wave == 0) {
        tdm_load_tile(&A [(size_t)bm * K], lds_addr_of(&sA[0][0]), K, TM, B_DIM);
        tdm_load_tile(&Bw[(size_t)bn * K], lds_addr_of(&sB[0][0]), K, TN, Nrows);
    }
    for (int kb = 0; kb < K; kb += KB) {
        const int cur = (kb >> 5) & 1;
        __builtin_amdgcn_s_wait_tensorcnt(0);   // DMA into buf[cur] complete
        __syncthreads();
        if (wave == 0 && kb + KB < K) {         // kick DMA for next tile
            tdm_load_tile(&A [(size_t)bm * K + kb + KB],
                          lds_addr_of(&sA[cur ^ 1][0]), K, TM, B_DIM);
            tdm_load_tile(&Bw[(size_t)bn * K + kb + KB],
                          lds_addr_of(&sB[cur ^ 1][0]), K, TN, Nrows);
        }
        const __bf16* cA = sA[cur];
        const __bf16* cB = sB[cur];
        FragBF a[4], b[4];
#pragma unroll
        for (int f = 0; f < 4; ++f) {
            // A frag: lane holds m=lane%16; K = {8q..8q+7, 8q+16..8q+23}
            const __bf16* pa = &cA[(m0 + f * 16 + lrow) * PITCH + half * 8];
            a[f].h[0] = *(const v8bf*)pa;
            a[f].h[1] = *(const v8bf*)(pa + 16);
            // B frag: lane holds n=lane%16; K = {8q..8q+7, 8q+16..8q+23}
            const __bf16* pb = &cB[(n0 + f * 16 + lrow) * PITCH + half * 8];
            b[f].h[0] = *(const v8bf*)pb;
            b[f].h[1] = *(const v8bf*)(pb + 16);
        }
#pragma unroll
        for (int fm = 0; fm < 4; ++fm)
#pragma unroll
            for (int fn = 0; fn < 4; ++fn)
                acc[fm][fn] = __builtin_amdgcn_wmma_f32_16x16x32_bf16(
                    false, a[fm].v, false, b[fn].v,
                    (short)0, acc[fm][fn], false, false);
    }
#else
    // ---- synchronous staging fallback ---------------------------------
    const int ldrow = tid >> 2;          // 0..63
    const int ldoff = (tid & 3) * 8;     // bf16 element offset within K-block
    for (int kb = 0; kb < K; kb += KB) {
        __syncthreads();
#pragma unroll
        for (int i = 0; i < 2; ++i) {
            int r = ldrow + i * 64;
            *(int4*)&sA[0][r * PITCH + ldoff] =
                *(const int4*)&A[(size_t)(bm + r) * K + kb + ldoff];
        }
#pragma unroll
        for (int i = 0; i < 4; ++i) {
            int r = ldrow + i * 64;
            *(int4*)&sB[0][r * PITCH + ldoff] =
                *(const int4*)&Bw[(size_t)(bn + r) * K + kb + ldoff];
        }
        if (kb + KB < K) {
            __builtin_prefetch(&A [(size_t)(bm + ldrow) * K + kb + KB + ldoff], 0, 1);
            __builtin_prefetch(&Bw[(size_t)(bn + ldrow) * K + kb + KB + ldoff], 0, 1);
        }
        __syncthreads();
        FragBF a[4], b[4];
#pragma unroll
        for (int f = 0; f < 4; ++f) {
            const __bf16* pa = &sA[0][(m0 + f * 16 + lrow) * PITCH + half * 8];
            a[f].h[0] = *(const v8bf*)pa;
            a[f].h[1] = *(const v8bf*)(pa + 16);
            const __bf16* pb = &sB[0][(n0 + f * 16 + lrow) * PITCH + half * 8];
            b[f].h[0] = *(const v8bf*)pb;
            b[f].h[1] = *(const v8bf*)(pb + 16);
        }
#pragma unroll
        for (int fm = 0; fm < 4; ++fm)
#pragma unroll
            for (int fn = 0; fn < 4; ++fn)
                acc[fm][fn] = __builtin_amdgcn_wmma_f32_16x16x32_bf16(
                    false, a[fm].v, false, b[fn].v,
                    (short)0, acc[fm][fn], false, false);
    }
#endif

    // Epilogue. C/D layout: lane holds n=lane%16, VGPR v holds m = v + 8*(lane/16).
#pragma unroll
    for (int fm = 0; fm < 4; ++fm) {
#pragma unroll
        for (int fn = 0; fn < 4; ++fn) {
#pragma unroll
            for (int v = 0; v < 8; ++v) {
                int row = bm + m0 + fm * 16 + half * 8 + v;
                int col = bn + n0 + fn * 16 + lrow;
                float g = acc[fm][fn][v] + bias[col];
                if (MODE == 0) {
                    size_t hrow = (size_t)row * H_DIM;
                    if (col < H_DIM) {                       // resetgate
                        float rg = sigmoidf(g);
                        rgh[hrow + col] = (__bf16)(hidden[hrow + col] * rg);
                    } else if (col < 2 * H_DIM) {            // inputgate
                        ig[hrow + (col - H_DIM)] = sigmoidf(g);
                    } else {                                  // i_n (pre-act)
                        inb[hrow + (col - 2 * H_DIM)] = g;
                    }
                } else {
                    size_t idx = (size_t)row * H_DIM + col;
                    float ng = sigmoidf(inb[idx] + g);       // g = h_n (incl. bh_n)
                    float h  = hidden[idx];
                    outp[idx] = h + (1.0f - h) * ig[idx] * ng;
                }
            }
        }
    }
}

// ---------------------------------------------------------------------------
extern "C" void kernel_launch(void* const* d_in, const int* in_sizes, int n_in,
                              void* d_out, int out_size, void* d_ws, size_t ws_size,
                              hipStream_t stream) {
    const float* x      = (const float*)d_in[0];
    const float* hidden = (const float*)d_in[1];
    const float* Wi     = (const float*)d_in[2];
    const float* bi     = (const float*)d_in[3];
    const float* Wh     = (const float*)d_in[4];
    const float* bh     = (const float*)d_in[5];
    float* out = (float*)d_out;
    char*  ws  = (char*)d_ws;

    size_t off = 0;
    __bf16* Xcat = (__bf16*)(ws + off); off += (size_t)B_DIM * 2048 * 2;        // 64 MB
    __bf16* Wcat = (__bf16*)(ws + off); off += (size_t)3072 * 2048 * 2;         // 12 MB
    __bf16* Whn  = (__bf16*)(ws + off); off += (size_t)H_DIM * H_DIM * 2;       //  2 MB
    float*  bcat = (float*) (ws + off); off += (size_t)3072 * 4 + 256;
    __bf16* rgh  = (__bf16*)(ws + off); off += (size_t)B_DIM * H_DIM * 2;       // 32 MB
    float*  ig   = (float*) (ws + off); off += (size_t)B_DIM * H_DIM * 4;       // 64 MB
    float*  inb  = (float*) (ws + off); off += (size_t)B_DIM * H_DIM * 4;       // 64 MB

    // 1) bf16 packing / fused weight construction
    prep_xcat<<<(B_DIM * 2048) / 256, 256, 0, stream>>>(x, hidden, Xcat);
    prep_wcat<<<(3072 * 2048) / 256, 256, 0, stream>>>(Wi, Wh, Wcat);
    prep_whn <<<(H_DIM * H_DIM) / 256, 256, 0, stream>>>(Wh, Whn);
    prep_bcat<<<3072 / 256, 256, 0, stream>>>(bi, bh, bcat);

    // 2) fused gate GEMM: [x|hidden](16384x2048) * Wcat^T(2048x3072)
    dim3 g1(B_DIM / TM, (3 * H_DIM) / TN);
    gru_gemm<0><<<g1, 256, 0, stream>>>(Xcat, Wcat, bcat, 2048, 3 * H_DIM,
                                        hidden, rgh, ig, inb, nullptr);

    // 3) dependent GEMM + final combine: (hidden*r)(16384x1024) * Wh_n^T
    dim3 g2(B_DIM / TM, H_DIM / TN);
    gru_gemm<1><<<g2, 256, 0, stream>>>(rgh, Whn, bh + 2 * H_DIM, 1024, H_DIM,
                                        hidden, nullptr, ig, inb, out);
}
